// GeGeLayer_55774445306519
// MI455X (gfx1250) — compile-verified, matrix operand order checked by reading, not run on hardware
//
#include <hip/hip_runtime.h>

#define BATCH   32
#define C_CH    16
#define IN_N    1728
#define N_FULL  1872   // padded / output n
#define N_TILES (N_FULL / 16)   // 117

typedef __attribute__((ext_vector_type(16))) __bf16 v16bf;
typedef __attribute__((ext_vector_type(8)))  float  v8f;

// ---------------------------------------------------------------------------
// Kernel 1: score[b,i] = sum_c xf[b,c,i]*w[c] + bias   (pad region -> bias)
// ---------------------------------------------------------------------------
__global__ void score_kernel(const float* __restrict__ x,
                             const float* __restrict__ w,
                             const float* __restrict__ bias,
                             float* __restrict__ scores) {
    int idx = blockIdx.x * blockDim.x + threadIdx.x;
    if (idx >= BATCH * N_FULL) return;
    int b = idx / N_FULL;
    int i = idx - b * N_FULL;
    float s = bias[0];
    if (i < IN_N) {
        const float* xb = x + (size_t)b * C_CH * IN_N + i;
#pragma unroll
        for (int c = 0; c < C_CH; ++c) s += xb[(size_t)c * IN_N] * w[c];
    }
    scores[idx] = s;
}

// ---------------------------------------------------------------------------
// Kernel 2: per-batch descending sort via bitonic sort of 2048 negated keys
// (+INF padding sorts to the tail). 1024 threads, 8KB LDS.
// ---------------------------------------------------------------------------
__global__ void sort_kernel(const float* __restrict__ scores,
                            float* __restrict__ sorted) {
    __shared__ float s[2048];
    int b = blockIdx.x;
    int tid = threadIdx.x;
    for (int i = tid; i < 2048; i += 1024)
        s[i] = (i < N_FULL) ? -scores[b * N_FULL + i] : __builtin_inff();
    __syncthreads();
    for (int k = 2; k <= 2048; k <<= 1) {
        for (int j = k >> 1; j > 0; j >>= 1) {
            for (int t = tid; t < 2048; t += 1024) {
                int ixj = t ^ j;
                if (ixj > t) {
                    bool up = ((t & k) == 0);
                    float a = s[t], c = s[ixj];
                    if ((a > c) == up) { s[t] = c; s[ixj] = a; }
                }
            }
            __syncthreads();
        }
    }
    for (int i = tid; i < N_FULL; i += 1024)
        sorted[b * N_FULL + i] = -s[i];
}

// ---------------------------------------------------------------------------
// Kernel 3: build 16 P-rows per block, write transposed probs_t[b][j][r].
// grid = (N_FULL/16, BATCH), 256 threads.
// ---------------------------------------------------------------------------
__global__ void prow_kernel(const float* __restrict__ scores,
                            const float* __restrict__ sorted,
                            float* __restrict__ probs) {
    __shared__ float sc[N_FULL];
    __shared__ float red1[256], red2[256];
    __shared__ float srt16[16], thr16[16], max16[16];
    int r0  = blockIdx.x * 16;
    int b   = blockIdx.y;
    int tid = threadIdx.x;

    const float* sb = scores + (size_t)b * N_FULL;
    for (int j = tid; j < N_FULL; j += 256) sc[j] = sb[j];
    if (tid < 16) srt16[tid] = sorted[(size_t)b * N_FULL + r0 + tid];
    __syncthreads();

    // 16 sequential block-wide top-2 reductions -> thr, rowmax
    for (int rr = 0; rr < 16; ++rr) {
        float srt_r = srt16[rr];
        float m1 = -__builtin_inff(), m2 = -__builtin_inff();
        for (int j = tid; j < N_FULL; j += 256) {
            float d = -fabsf(sc[j] - srt_r);
            if (d > m1)      { m2 = m1; m1 = d; }
            else if (d > m2) { m2 = d; }
        }
        red1[tid] = m1; red2[tid] = m2;
        __syncthreads();
        for (int off = 128; off > 0; off >>= 1) {
            if (tid < off) {
                float a1 = red1[tid], a2 = red2[tid];
                float b1 = red1[tid + off], b2 = red2[tid + off];
                red1[tid] = fmaxf(a1, b1);
                red2[tid] = fmaxf(fminf(a1, b1), fmaxf(a2, b2));
            }
            __syncthreads();
        }
        if (tid == 0) {
            float thr = 0.5f * (red1[0] + red2[0]);
            thr16[rr] = thr;
            max16[rr] = red1[0] - thr;  // = max_j relu(D - thr)
        }
        __syncthreads();
    }

    // write phase: tid%16 -> r (contiguous 64B segments), tid/16 -> j lane
    float* pb = probs + (size_t)b * N_FULL * N_FULL;
    int rl = tid & 15;
    int jl = tid >> 4;                 // 0..15
    float srt_r = srt16[rl], thr = thr16[rl], mx = max16[rl];
    for (int jb = 0; jb < N_FULL; jb += 16) {
        int j = jb + jl;
        float d = -fabsf(sc[j] - srt_r);
        float p = fmaxf(d - thr, 0.0f) / mx;   // matches jnp P / max (0/0 -> NaN)
        pb[(size_t)j * N_FULL + (r0 + rl)] = p;
    }
}

// ---------------------------------------------------------------------------
// Kernel 4: y[b,c,r] = sum_j P[b,r,j] * xf[b,c,j] via v_wmma_f32_16x16x32_bf16
// 4 waves per block, one 16(M=r) x 16(N=c) tile per wave.
// Since xf[c,j] == 0 for j >= IN_N (zero padding) and IN_N % 32 == 0, the K
// loop covers exactly [0, IN_N) in 54 full chunks -> NO bounds checks, so
// loads are unpredicated and can be clustered ahead of the cvt/pack sequence.
// ---------------------------------------------------------------------------
__global__ void __launch_bounds__(128)
y_wmma_kernel(const float* __restrict__ x,
              const float* __restrict__ probs,
              float* __restrict__ y) {
    int lane  = threadIdx.x & 31;
    int wave  = threadIdx.x >> 5;
    int tile  = blockIdx.x * 4 + wave;           // r-tile index
    if (tile >= N_TILES) return;                 // uniform per wave
    int r0    = tile * 16;
    int b     = blockIdx.y;
    int m     = lane & 15;       // A row (r) / B col (c) / D col (c)
    int khalf = lane >> 4;       // 0: K base 0, 1: K base +8

    const float* pcol = probs + (size_t)b * N_FULL * N_FULL + (r0 + m); // column r0+m
    const float* xb   = x + (size_t)b * C_CH * IN_N + (size_t)m * IN_N; // channel m

    v8f acc = {};
    for (int j0 = 0; j0 < IN_N; j0 += 32) {
        v16bf av, bv;
#pragma unroll
        for (int v = 0; v < 8; ++v) {
            // 16-bit A/B fragment layout: VGPR v holds K pair {k0, k0+1}
            int k0 = (v < 4 ? 2 * v : 16 + 2 * (v - 4)) + 8 * khalf;
            int ja = j0 + k0;
            float  a0 = pcol[(size_t)ja * N_FULL];
            float  a1 = pcol[(size_t)(ja + 1) * N_FULL];
            float2 bp = *(const float2*)(xb + ja);     // ja even -> 8B aligned
            av[2 * v] = (__bf16)a0; av[2 * v + 1] = (__bf16)a1;
            bv[2 * v] = (__bf16)bp.x; bv[2 * v + 1] = (__bf16)bp.y;
        }
        acc = __builtin_amdgcn_wmma_f32_16x16x32_bf16(
            /*neg_a=*/false, av, /*neg_b=*/false, bv,
            /*c_mod=*/(short)0, acc, /*reuse_a=*/false, /*reuse_b=*/false);
    }

    // D layout: lane<16 -> M=v, lane>=16 -> M=v+8; N = lane&15 = channel
    float* yb = y + (size_t)b * C_CH * N_FULL + (size_t)m * N_FULL;
#pragma unroll
    for (int v = 0; v < 8; ++v)
        yb[r0 + v + 8 * khalf] = acc[v];
}

// ---------------------------------------------------------------------------
extern "C" void kernel_launch(void* const* d_in, const int* in_sizes, int n_in,
                              void* d_out, int out_size, void* d_ws, size_t ws_size,
                              hipStream_t stream) {
    const float* x    = (const float*)d_in[0];
    const float* w    = (const float*)d_in[1];
    const float* bias = (const float*)d_in[2];

    float* out   = (float*)d_out;
    float* y     = out;                                      // (32,16,1872)
    float* probs = out + (size_t)BATCH * C_CH * N_FULL;      // probs_t (32,1872,1872)

    float* scores = (float*)d_ws;                            // (32,1872)
    float* sorted = scores + BATCH * N_FULL;                 // (32,1872)

    int tot = BATCH * N_FULL;
    score_kernel<<<(tot + 255) / 256, 256, 0, stream>>>(x, w, bias, scores);
    sort_kernel<<<BATCH, 1024, 0, stream>>>(scores, sorted);

    dim3 g3(N_TILES, BATCH);
    prow_kernel<<<g3, 256, 0, stream>>>(scores, sorted, probs);

    dim3 g4((N_TILES + 3) / 4, BATCH);
    y_wmma_kernel<<<g4, 128, 0, stream>>>(x, probs, y);
}
